// GCN0000_20469814133394
// MI455X (gfx1250) — matrix operand order repaired and checked by reading, not compiled, not run on hardware
//
#include <hip/hip_runtime.h>
#include <hip/hip_bf16.h>

// ---------------------------------------------------------------------------
// 2-layer GCN for MI455X (gfx1250): bf16 WMMA GEMMs + atomic edge propagation
// ---------------------------------------------------------------------------

typedef __attribute__((ext_vector_type(16))) __bf16 v16bf;
typedef __attribute__((ext_vector_type(8)))  float  v8f;

#define NN   50000
#define FIN  512
#define FHID 128
#define FOUT 64

static __device__ __forceinline__ __bf16 tobf(float f) {
  return static_cast<__bf16>(f);
}

// ---------------------------------------------------------------------------
// Pack a row-major K x N fp32 weight matrix into bf16 WMMA B-fragments.
// Fragment layout (ISA 7.12.2, 16-bit B KxN): per (n-tile j, k-chunk t):
//   lanes 0..15  : n = j*16+lane,     elements i=0..7 -> K = t*32 + i,     i=8..15 -> K = t*32 + 16 + (i-8)
//   lanes 16..31 : n = j*16+lane-16,  elements i=0..7 -> K = t*32 + 8 + i, i=8..15 -> K = t*32 + 24 + (i-8)
// Stored contiguously: Bp[((j*KC + t)*32 + lane)*16 + i]
// ---------------------------------------------------------------------------
__global__ void k_pack_b(const float* __restrict__ B, __bf16* __restrict__ Bp,
                         int K, int N) {
  int idx = blockIdx.x * blockDim.x + threadIdx.x;
  if (idx >= K * N) return;
  int i    = idx & 15;
  int lane = (idx >> 4) & 31;
  int ft   = idx >> 9;            // fragment index = j*KC + t
  int KC   = K >> 5;
  int j    = ft / KC;
  int t    = ft - j * KC;
  int n    = j * 16 + (lane & 15);
  int k    = t * 32 + ((i >> 3) << 4) + ((lane >> 4) << 3) + (i & 7);
  Bp[idx]  = tobf(B[k * N + n]);
}

// ---------------------------------------------------------------------------
// C[M x (NT*16)] = A[M x (KC*32)] (fp32, converted to bf16) @ Bp (packed bf16)
// One wave per 16-row M-tile; NT accumulators of 8 VGPRs each; A fragment
// reused across all NT column tiles. EXEC is all-1s through every WMMA.
// ---------------------------------------------------------------------------
template <int NT, int KC>
__global__ __launch_bounds__(256)
void k_gemm_wmma(const float* __restrict__ A, const __bf16* __restrict__ Bp,
                 float* __restrict__ C, int mTiles) {
  const int N    = NT * 16;
  const int K    = KC * 32;
  const int lane = threadIdx.x & 31;
  int tile       = blockIdx.x * 8 + (threadIdx.x >> 5);
  const bool valid = (tile < mTiles);
  if (!valid) tile = mTiles - 1;           // clamp: keep EXEC all-1s for WMMA

  const int m          = tile * 16 + (lane & 15);
  const float* arow    = A + (size_t)m * K;
  const int khalf      = (lane >> 4) * 8;  // lanes 0-15 -> K+0, lanes 16-31 -> K+8

  v8f zero = {0.f, 0.f, 0.f, 0.f, 0.f, 0.f, 0.f, 0.f};
  v8f acc[NT];
#pragma unroll
  for (int j = 0; j < NT; ++j) acc[j] = zero;

  for (int kc = 0; kc < KC; ++kc) {
    const int k0 = kc * 32 + khalf;
    // A fragment: elements 0..7 -> K = k0..k0+7 ; 8..15 -> K = k0+16..k0+23
    float4 f0 = *(const float4*)(arow + k0);
    float4 f1 = *(const float4*)(arow + k0 + 4);
    float4 f2 = *(const float4*)(arow + k0 + 16);
    float4 f3 = *(const float4*)(arow + k0 + 20);
    __builtin_prefetch(arow + k0 + 32, 0, 0);   // global_prefetch_b8
    v16bf a;
    a[0]  = tobf(f0.x); a[1]  = tobf(f0.y); a[2]  = tobf(f0.z); a[3]  = tobf(f0.w);
    a[4]  = tobf(f1.x); a[5]  = tobf(f1.y); a[6]  = tobf(f1.z); a[7]  = tobf(f1.w);
    a[8]  = tobf(f2.x); a[9]  = tobf(f2.y); a[10] = tobf(f2.z); a[11] = tobf(f2.w);
    a[12] = tobf(f3.x); a[13] = tobf(f3.y); a[14] = tobf(f3.z); a[15] = tobf(f3.w);
#pragma unroll
    for (int j = 0; j < NT; ++j) {
      v16bf b = *(const v16bf*)(Bp + (((size_t)(j * KC + kc) * 32 + lane) << 4));
      acc[j] = __builtin_amdgcn_wmma_f32_16x16x32_bf16(
          /*neg_a=*/false, a, /*neg_b=*/false, b,
          /*c_mod=*/(short)0, acc[j], /*reuse_a=*/false, /*reuse_b=*/false);
    }
  }

  if (valid) {
    // C/D layout: lanes 0-15: M = r, N = lane ; lanes 16-31: M = r+8, N = lane-16
    const int rbase = tile * 16 + (lane >> 4) * 8;
    const int nb    = lane & 15;
#pragma unroll
    for (int j = 0; j < NT; ++j) {
#pragma unroll
      for (int r = 0; r < 8; ++r) {
        C[(size_t)(rbase + r) * N + j * 16 + nb] = acc[j][r];
      }
    }
  }
}

// --------------------------- graph plumbing --------------------------------

__global__ void k_deg_init(float* __restrict__ deg, int n) {
  int i = blockIdx.x * blockDim.x + threadIdx.x;
  if (i < n) deg[i] = 1.0f;                 // self-loop contribution
}

__global__ void k_deg_count(const int* __restrict__ dst, float* __restrict__ deg, int E) {
  int e = blockIdx.x * blockDim.x + threadIdx.x;
  if (e < E) atomicAdd(&deg[dst[e]], 1.0f);
}

__global__ void k_dinv(const float* __restrict__ deg, float* __restrict__ dinv, int n) {
  int i = blockIdx.x * blockDim.x + threadIdx.x;
  if (i < n) {
    float d = deg[i];
    dinv[i] = (d > 0.f) ? rsqrtf(d) : 0.f;
  }
}

__global__ void k_fill_bias(float* __restrict__ P, const float* __restrict__ b,
                            int mask, int total) {
  int i = blockIdx.x * blockDim.x + threadIdx.x;
  if (i < total) P[i] = b[i & mask];
}

// Edge-parallel propagate: out[dst] += h[src] * dinv[src]*dinv[dst]
// Virtual edges e in [E, E+NN) are self loops. float4 per thread.
__global__ void k_prop(const float* __restrict__ H, float* __restrict__ P,
                       const int* __restrict__ src, const int* __restrict__ dst,
                       const float* __restrict__ dinv, int E, int f4shift, int total) {
  int tid = blockIdx.x * blockDim.x + threadIdx.x;
  if (tid >= total) return;
  int c = tid & ((1 << f4shift) - 1);
  int e = tid >> f4shift;
  int s, d;
  if (e < E) { s = src[e]; d = dst[e]; }
  else       { s = e - E;  d = s; }
  float w = dinv[s] * dinv[d];
  float4 h = ((const float4*)H)[((size_t)s << f4shift) + c];
  float* p = P + ((((size_t)d << f4shift) + c) << 2);
  atomicAdd(p + 0, h.x * w);
  atomicAdd(p + 1, h.y * w);
  atomicAdd(p + 2, h.z * w);
  atomicAdd(p + 3, h.w * w);
}

__global__ void k_relu(float* __restrict__ P, int total) {
  int i = blockIdx.x * blockDim.x + threadIdx.x;
  if (i < total) P[i] = fmaxf(P[i], 0.f);
}

// One wave per node: 64 features -> 2 per lane (wave32); fused +bl.
__global__ __launch_bounds__(256)
void k_logsoftmax(const float* __restrict__ Fm, const float* __restrict__ bl,
                  float* __restrict__ out, int n) {
  int node = blockIdx.x * 8 + (threadIdx.x >> 5);
  int lane = threadIdx.x & 31;
  if (node >= n) return;
  float v0 = Fm[(size_t)node * 64 + lane]      + bl[lane];
  float v1 = Fm[(size_t)node * 64 + 32 + lane] + bl[32 + lane];
  float m = fmaxf(v0, v1);
  for (int off = 16; off > 0; off >>= 1) m = fmaxf(m, __shfl_xor(m, off, 32));
  float s = __expf(v0 - m) + __expf(v1 - m);
  for (int off = 16; off > 0; off >>= 1) s += __shfl_xor(s, off, 32);
  float l = m + __logf(s);
  out[(size_t)node * 64 + lane]      = v0 - l;
  out[(size_t)node * 64 + 32 + lane] = v1 - l;
}

// ------------------------------- driver ------------------------------------

extern "C" void kernel_launch(void* const* d_in, const int* in_sizes, int n_in,
                              void* d_out, int out_size, void* d_ws, size_t ws_size,
                              hipStream_t stream) {
  const float* x  = (const float*)d_in[0];
  const int*   ei = (const int*)  d_in[1];   // [2, E] int32
  const float* W1 = (const float*)d_in[2];
  const float* b1 = (const float*)d_in[3];
  const float* W2 = (const float*)d_in[4];
  const float* b2 = (const float*)d_in[5];
  const float* Wl = (const float*)d_in[6];
  const float* bl = (const float*)d_in[7];
  float* out = (float*)d_out;

  const int E = in_sizes[1] / 2;
  const int* src = ei;
  const int* dst = ei + E;

  // ---- workspace carving (256B aligned) ----
  size_t o = 0;
  auto carve = [&](size_t bytes) -> char* {
    char* p = (char*)d_ws + o;
    o += (bytes + 255) & ~(size_t)255;
    return p;
  };
  float*  H1   = (float*) carve((size_t)NN * FHID * 4);  // also reused for Fm
  float*  P1   = (float*) carve((size_t)NN * FHID * 4);
  float*  H2   = (float*) carve((size_t)NN * FOUT * 4);
  float*  P2   = (float*) carve((size_t)NN * FOUT * 4);
  __bf16* W1p  = (__bf16*)carve((size_t)FIN  * FHID * 2);
  __bf16* W2p  = (__bf16*)carve((size_t)FHID * FOUT * 2);
  __bf16* Wlp  = (__bf16*)carve((size_t)FOUT * FOUT * 2);
  float*  deg  = (float*) carve((size_t)NN * 4);
  float*  dinv = (float*) carve((size_t)NN * 4);
  float*  Fm   = H1;  // H1 dead after propagate-1

  const int TB = 256;
  const int mTiles = NN / 16;                       // 3125
  const int gemmBlocks = (mTiles + 7) / 8;          // 8 waves / block

  // ---- pack weights into WMMA bf16 B-fragment layout ----
  k_pack_b<<<(FIN * FHID + TB - 1) / TB, TB, 0, stream>>>(W1, W1p, FIN, FHID);
  k_pack_b<<<(FHID * FOUT + TB - 1) / TB, TB, 0, stream>>>(W2, W2p, FHID, FOUT);
  k_pack_b<<<(FOUT * FOUT + TB - 1) / TB, TB, 0, stream>>>(Wl, Wlp, FOUT, FOUT);

  // ---- symmetric normalization coefficients ----
  k_deg_init<<<(NN + TB - 1) / TB, TB, 0, stream>>>(deg, NN);
  k_deg_count<<<(E + TB - 1) / TB, TB, 0, stream>>>(dst, deg, E);
  k_dinv<<<(NN + TB - 1) / TB, TB, 0, stream>>>(deg, dinv, NN);

  // ---- layer 1: H1 = X @ W1 ; P1 = b1 + scatter(norm * H1) ; relu ----
  k_gemm_wmma<FHID / 16, FIN / 32><<<gemmBlocks, TB, 0, stream>>>(x, W1p, H1, mTiles);
  k_fill_bias<<<((NN * FHID) + TB - 1) / TB, TB, 0, stream>>>(P1, b1, FHID - 1, NN * FHID);
  {
    int total = (E + NN) * (FHID / 4);
    k_prop<<<(total + TB - 1) / TB, TB, 0, stream>>>(H1, P1, src, dst, dinv, E, 5, total);
  }
  k_relu<<<((NN * FHID) + TB - 1) / TB, TB, 0, stream>>>(P1, NN * FHID);

  // ---- layer 2: H2 = relu(P1) @ W2 ; P2 = b2 + scatter(norm * H2) ----
  k_gemm_wmma<FOUT / 16, FHID / 32><<<gemmBlocks, TB, 0, stream>>>(P1, W2p, H2, mTiles);
  k_fill_bias<<<((NN * FOUT) + TB - 1) / TB, TB, 0, stream>>>(P2, b2, FOUT - 1, NN * FOUT);
  {
    int total = (E + NN) * (FOUT / 4);
    k_prop<<<(total + TB - 1) / TB, TB, 0, stream>>>(H2, P2, src, dst, dinv, E, 4, total);
  }

  // ---- head: Fm = P2 @ Wl ; out = log_softmax(Fm + bl) ----
  k_gemm_wmma<FOUT / 16, FOUT / 32><<<gemmBlocks, TB, 0, stream>>>(P2, Wlp, Fm, mTiles);
  k_logsoftmax<<<(NN + 7) / 8, TB, 0, stream>>>(Fm, bl, out, NN);
}
